// SpatialEmbLoss_71863392796685
// MI455X (gfx1250) — matrix operand order repaired and checked by reading, not compiled, not run on hardware
//
#include <hip/hip_runtime.h>
#include <hip/hip_bf16.h>

// ---------------- problem constants ----------------
#define BATCH   2
#define HDIM    1024
#define WDIM    1024
#define HW      (HDIM * WDIM)        // 1,048,576
#define KMAX    16
#define NBINS   1024
#define CHUNKS  256                  // blocks along pixel dim for k_hist
#define P1BLKS  1024                 // blocks along pixel dim for k_phase1

// ---------------- workspace layout (in dwords) ----------------
#define WS_SUMS    0                         // [B][16 feat][16 inst] f32 = 512
#define WS_SEEDBG  512                       // [B] f32
#define WS_SEEDFG  514                       // [B][16] f32
#define WS_PARAMS  546                       // [B][16][8] f32 = 256
#define WS_HPOS    1024                      // [B][16][NBINS] u32 = 32768
#define WS_HNEG    (WS_HPOS + BATCH*KMAX*NBINS)     // 33792
#define WS_TOTAL   (WS_HNEG + BATCH*KMAX*NBINS)     // 66560 dwords ~ 260 KB

typedef __attribute__((ext_vector_type(16))) _Float16 v16h;
typedef __attribute__((ext_vector_type(8)))  _Float16 v8h;
typedef __attribute__((ext_vector_type(8)))  float    v8f;

__device__ __forceinline__ void atomicAddF(float* p, float v) {
  __hip_atomic_fetch_add(p, v, __ATOMIC_RELAXED, __HIP_MEMORY_SCOPE_AGENT);
}

// fast (1-ulp rcp based) transcendentals -- fine for a scalar loss
__device__ __forceinline__ float fastSigmoid(float x) {
  return __builtin_amdgcn_rcpf(1.0f + __expf(-x));           // v_rcp_f32 + v_exp_f32
}
__device__ __forceinline__ float fastTanh(float x) {
  const float t = __expf(2.0f * x);                          // inf / 0 saturate to +/-1
  return 1.0f - 2.0f * __builtin_amdgcn_rcpf(t + 1.0f);
}

// wave-local LDS publish/consume fence: LDS ops within a wave are in-order
// (ISA 7.3), so we only need to stop the compiler reordering across the point.
__device__ __forceinline__ void waveLdsFence() {
  asm volatile("" ::: "memory");
  __builtin_amdgcn_wave_barrier();
  asm volatile("" ::: "memory");
}

// ---------------- zero the workspace (harness poisons it) ----------------
__global__ void k_zero(float* ws, int n) {
  int i = blockIdx.x * 256 + threadIdx.x;
  if (i < n) ws[i] = 0.0f;
}

// ---------------- phase 1: mask-weighted feature sums via WMMA ----------------
// A (16x32, f16): rows = features {1, xm, ym, cen, cen*xm, cen*ym, s0, s1, s0^2, s1^2, 0...}
//   staged in LDS as f16 [feat][pixel] so a lane's fragment = 2x ds_load_b128.
// B (32x16, f16): B[pixel][k] = (inst == k+1); lane reads 16 contiguous ids (4x int4).
// C/D (16x16, f32): per-wave accumulators -> LDS ds_add_f32 -> one global atomic/thread.
__global__ __launch_bounds__(256) void k_phase1(
    const float* __restrict__ pred, const float* __restrict__ xym,
    const int* __restrict__ inst, const int* __restrict__ lab,
    const unsigned char* __restrict__ cen, float* __restrict__ ws)
{
  const int b    = blockIdx.z;
  const int tid  = threadIdx.x;
  const int lane = tid & 31;
  const int wave = tid >> 5;
  const int m    = lane & 15;       // feature row (A) / instance col (B)
  const int hi   = lane >> 4;       // lane half selects K sub-block

  __shared__ _Float16 aSh[8][16][32];  // [wave][feat][pixel]  8 KB, per-wave region
  __shared__ int      iS[8][32];       // [wave][pixel] instance id
  __shared__ float    csum[256];       // block-level 16x16 accumulator
  __shared__ float    red[256];

  csum[tid] = 0.0f;
  __syncthreads();

  const float* predB = pred + (size_t)b * 5 * HW;
  const int*   instB = inst + (size_t)b * HW;
  const int*   labB  = lab  + (size_t)b * HW;
  const unsigned char* cenB = cen + (size_t)b * HW;

  const int base = blockIdx.x * (HW / P1BLKS);   // 1024 pixels per block

  v8f   acc = {};
  float sbg = 0.0f;

#pragma unroll
  for (int it = 0; it < 4; ++it) {
    const int p = base + wave * 128 + it * 32 + lane;   // pixel in [0, HW)

    const float xm = xym[p];
    const float ym = xym[HW + p];
    const float s0 = predB[2 * HW + p];
    const float s1 = predB[3 * HW + p];
    const float seed = fastSigmoid(predB[4 * HW + p]);
    const int   iv = instB[p];
    const float cw = cenB[p] ? 1.0f : 0.0f;
    sbg += (labB[p] == 0) ? seed * seed : 0.0f;   // branchless, EXEC stays full

    const float f[16] = {1.0f, xm, ym, cw, cw * xm, cw * ym,
                         s0, s1, s0 * s0, s1 * s1,
                         0.f, 0.f, 0.f, 0.f, 0.f, 0.f};

    waveLdsFence();                  // prior iteration's fragment reads retired order-wise
#pragma unroll
    for (int r = 0; r < 16; ++r) aSh[wave][r][lane] = (_Float16)f[r];
    iS[wave][lane] = iv;
    waveLdsFence();                  // publish before cross-lane consume

    // A fragment: feature row m, pixels {16hi..16hi+7} then {16hi+8..16hi+15}
    //   -> two contiguous 16B rows, independent b128 loads (one dscnt wait).
    const v8h alo = *(const v8h*)&aSh[wave][m][16 * hi];
    const v8h ahi = *(const v8h*)&aSh[wave][m][16 * hi + 8];
    v16h A;
#pragma unroll
    for (int e = 0; e < 8; ++e) { A[e] = alo[e]; A[8 + e] = ahi[e]; }

    // B fragment: element e = one-hot(inst[pixel e+16hi] == m+1), e = 0..15
    const int4* ib = (const int4*)&iS[wave][16 * hi];
    const int4 i0 = ib[0], i1 = ib[1], i2 = ib[2], i3 = ib[3];
    const int iv16[16] = {i0.x, i0.y, i0.z, i0.w, i1.x, i1.y, i1.z, i1.w,
                          i2.x, i2.y, i2.z, i2.w, i3.x, i3.y, i3.z, i3.w};
    v16h Bv;
#pragma unroll
    for (int e = 0; e < 16; ++e)
      Bv[e] = (_Float16)((iv16[e] == m + 1) ? 1.0f : 0.0f);

    acc = __builtin_amdgcn_wmma_f32_16x16x32_f16(
        /*neg_a=*/false, A, /*neg_b=*/false, Bv,
        /*c_mod=*/(short)0, acc, /*reuse_a=*/false, /*reuse_b=*/false);
  }

  // stage 1: reduce the 8 waves' accumulators in LDS (ds_add_f32)
  // lane L vgpr v -> element row (v + 8*hi), col m
#pragma unroll
  for (int v = 0; v < 8; ++v) {
    const int row = v + 8 * hi;
    atomicAdd(&csum[row * 16 + m], acc[v]);
  }
  __syncthreads();
  // stage 2: one global atomic per thread (8x fewer than per-wave merge)
  atomicAddF(&ws[WS_SUMS + b * 256 + tid], csum[tid]);

  // seed_bg block reduction
  red[tid] = sbg;
  __syncthreads();
  for (int s = 128; s > 0; s >>= 1) {
    if (tid < s) red[tid] += red[tid + s];
    __syncthreads();
  }
  if (tid == 0) atomicAddF(&ws[WS_SEEDBG + b], red[0]);
}

// ---------------- derive per-instance parameters ----------------
__global__ void k_params(float* __restrict__ ws) {
  const int t = threadIdx.x;
  const int b = t >> 4, k = t & 15;
  const float* S = ws + WS_SUMS + b * 256;
  const float count = S[0 * 16 + k], sxm = S[1 * 16 + k], sym = S[2 * 16 + k];
  const float ccnt  = S[3 * 16 + k], cxm = S[4 * 16 + k], cym = S[5 * 16 + k];
  const float ss0   = S[6 * 16 + k], ss1 = S[7 * 16 + k];
  const float sq0   = S[8 * 16 + k], sq1 = S[9 * 16 + k];

  const float present = (count > 0.0f) ? 1.0f : 0.0f;
  const float cnt = fmaxf(count, 1.0f);
  const float mx = sxm / cnt, my = sym / cnt;
  const bool  onecen = (ccnt == 1.0f);
  const float cx = onecen ? cxm : mx;
  const float cy = onecen ? cym : my;
  const float sm0 = ss0 / cnt, sm1 = ss1 / cnt;
  const float var = ((sq0 - 2.0f * sm0 * ss0 + sm0 * sm0 * count) +
                     (sq1 - 2.0f * sm1 * ss1 + sm1 * sm1 * count)) / (2.0f * cnt);

  float* P = ws + WS_PARAMS + (b * 16 + k) * 8;
  P[0] = cx; P[1] = cy;
  P[2] = expf(10.0f * sm0);
  P[3] = expf(10.0f * sm1);
  P[4] = present; P[5] = var; P[6] = count; P[7] = 0.0f;
}

// ---------------- phase 2: error histograms (counting sort) + seed_fg ----------------
// One packed LDS atomic per pixel: pos count in [31:16], neg count in [15:0]
// (per-block per-bin count <= 4096 < 2^16, no overflow).
__global__ __launch_bounds__(256) void k_hist(
    const float* __restrict__ pred, const float* __restrict__ xym,
    const int* __restrict__ inst, float* __restrict__ ws)
{
  const int b = blockIdx.z, k = blockIdx.y;
  const float* P = ws + WS_PARAMS + (b * 16 + k) * 8;
  if (P[4] == 0.0f) return;                   // instance absent: loss masked out
  const float cx = P[0], cy = P[1], s0 = P[2], s1 = P[3];

  __shared__ unsigned int hcnt[NBINS];
  __shared__ float        red[256];
  for (int i = threadIdx.x; i < NBINS; i += 256) hcnt[i] = 0u;
  __syncthreads();

  const float* predB = pred + (size_t)b * 5 * HW;
  const int*   instB = inst + (size_t)b * HW;
  const int    perBlock = HW / CHUNKS;        // 4096
  const int    base = blockIdx.x * perBlock;

  float fg = 0.0f;
  for (int i = threadIdx.x; i < perBlock; i += 256) {
    const int p = base + i;
    // stream-ahead prefetch (speculative; OOB prefetches are dropped)
    __builtin_prefetch(&predB[p + 4096], 0, 0);
    __builtin_prefetch(&predB[HW + p + 4096], 0, 0);

    const float ex = fastTanh(predB[p])      + xym[p];
    const float ey = fastTanh(predB[HW + p]) + xym[HW + p];
    const float dx = ex - cx, dy = ey - cy;
    const float d  = __expf(-(s0 * dx * dx + s1 * dy * dy));
    const bool  pos = (instB[p] == k + 1);
    const float err = pos ? (2.0f - 2.0f * d) : (2.0f * d);
    if (pos) {
      const float t = fastSigmoid(predB[4 * HW + p]) - d;
      fg += t * t;
    }
    int bin = (int)(err * (NBINS * 0.5f));
    bin = min(max(bin, 0), NBINS - 1);
    atomicAdd(&hcnt[bin], pos ? 0x10000u : 1u);
  }
  __syncthreads();

  unsigned int* gHp = (unsigned int*)ws + WS_HPOS + (b * 16 + k) * NBINS;
  unsigned int* gHn = (unsigned int*)ws + WS_HNEG + (b * 16 + k) * NBINS;
  for (int i = threadIdx.x; i < NBINS; i += 256) {
    const unsigned int c = hcnt[i];
    if (c) {
      if (c >> 16)     atomicAdd(&gHp[i], c >> 16);
      if (c & 0xFFFFu) atomicAdd(&gHn[i], c & 0xFFFFu);
    }
  }

  red[threadIdx.x] = fg;
  __syncthreads();
  for (int s = 128; s > 0; s >>= 1) {
    if (threadIdx.x < s) red[threadIdx.x] += red[threadIdx.x + s];
    __syncthreads();
  }
  if (threadIdx.x == 0 && red[0] != 0.0f)
    atomicAddF(&ws[WS_SEEDFG + b * 16 + k], red[0]);
}

// ---------------- final: Lovasz from histograms + loss assembly ----------------
__global__ void k_final(float* __restrict__ ws, float* __restrict__ out) {
  __shared__ float lov[32], pres[32], varr[32], fgv[32];
  __shared__ float sampleLoss[2];
  const int t = threadIdx.x;

  {
    const int b = t >> 4, k = t & 15;
    const float* P = ws + WS_PARAMS + (b * 16 + k) * 8;
    const float Pn = P[6];                    // total positives
    const unsigned int* gHp = (const unsigned int*)ws + WS_HPOS + (b * 16 + k) * NBINS;
    const unsigned int* gHn = (const unsigned int*)ws + WS_HNEG + (b * 16 + k) * NBINS;

    float loss = 0.0f, jprev = 0.0f, n = 0.0f, pp = 0.0f;
    for (int bin = NBINS - 1; bin >= 0; --bin) {   // descending error order
      const float np = (float)gHp[bin];
      const float nn = (float)gHn[bin];
      const float tot = np + nn;
      if (tot > 0.0f) {
        n += tot; pp += np;
        const float uni  = Pn + n - pp;            // >= 1 whenever tot>0
        const float jacc = 1.0f - (Pn - pp) / uni;
        const float e    = (bin + 0.5f) * (2.0f / NBINS);   // bin-center error
        loss += e * (jacc - jprev);                // telescoped grad * relu(err)
        jprev = jacc;
      }
    }
    lov[t]  = loss;
    pres[t] = P[4];
    varr[t] = P[5];
    fgv[t]  = ws[WS_SEEDFG + b * 16 + k];
  }
  __syncthreads();

  if (t < 2) {
    const int b = t;
    float sp = 0.f, si = 0.f, sv = 0.f, sf = 0.f;
    for (int k = 0; k < 16; ++k) {
      const int i = b * 16 + k;
      sp += pres[i];
      si += lov[i] * pres[i];
      sv += varr[i] * pres[i];
      sf += fgv[i] * pres[i];
    }
    const float obj = fmaxf(sp, 1.0f);
    const float inst_loss = si / obj;
    const float var_loss  = sv / obj;
    const float seed_loss = (ws[WS_SEEDBG + b] + 1.0f /*FG_W*/ * sf) / (float)HW;
    sampleLoss[b] = 1.0f * inst_loss + 10.0f * var_loss + 1.0f * seed_loss;
  }
  __syncthreads();
  if (t == 0) out[0] = 0.5f * (sampleLoss[0] + sampleLoss[1]);
}

// ---------------- host launcher ----------------
extern "C" void kernel_launch(void* const* d_in, const int* in_sizes, int n_in,
                              void* d_out, int out_size, void* d_ws, size_t ws_size,
                              hipStream_t stream) {
  const float*         pred = (const float*)d_in[0];          // (B,5,H,W) f32
  const float*         xym  = (const float*)d_in[1];          // (2,H,W)  f32
  const int*           inst = (const int*)d_in[2];            // (B,H,W)  i32
  const int*           lab  = (const int*)d_in[3];            // (B,H,W)  i32
  const unsigned char* cen  = (const unsigned char*)d_in[4];  // (B,H,W)  bool
  float* ws  = (float*)d_ws;
  float* out = (float*)d_out;

  k_zero  <<<(WS_TOTAL + 255) / 256, 256, 0, stream>>>(ws, WS_TOTAL);
  k_phase1<<<dim3(P1BLKS, 1, BATCH), 256, 0, stream>>>(pred, xym, inst, lab, cen, ws);
  k_params<<<1, 32, 0, stream>>>(ws);
  k_hist  <<<dim3(CHUNKS, KMAX, BATCH), 256, 0, stream>>>(pred, xym, inst, ws);
  k_final <<<1, 32, 0, stream>>>(ws, out);
}
